// GraphTokenizer_47055661695548
// MI455X (gfx1250) — compile-verified, matrix operand order checked by reading, not run on hardware
//
#include <hip/hip_runtime.h>
#include <hip/hip_bf16.h>

#define NN 20000
#define DD 512
#define KK 3
#define MM 4096
#define EE 640000

typedef __attribute__((ext_vector_type(16))) __bf16 v16bf;
typedef __attribute__((ext_vector_type(8)))  __bf16 v8bf;
typedef __attribute__((ext_vector_type(8)))  float  v8f;
typedef __attribute__((ext_vector_type(4)))  int    v4i;

#if defined(__has_builtin)
#  if __has_builtin(__builtin_amdgcn_global_load_async_to_lds_b128)
#    define HAVE_ASYNC_LDS 1
#  endif
#endif
#ifndef HAVE_ASYNC_LDS
#  define HAVE_ASYNC_LDS 0
#endif

static __device__ inline void wait_asynccnt0() {
#if defined(__has_builtin) && __has_builtin(__builtin_amdgcn_s_wait_asynccnt)
  __builtin_amdgcn_s_wait_asynccnt(0);
#else
  asm volatile("s_wait_asynccnt 0x0" ::: "memory");
#endif
}

// ---------- fragment loaders (ISA 7.12.2 16-bit layouts, wave32) ----------
// A 16x32 bf16: lane L holds row M=L&15; K = (L>>4)*8 + {0..7} then 16+(L>>4)*8 + {0..7}
// B 32x16 bf16 (stored transposed, row = output column): same K packing.
static __device__ inline v16bf load_frag_row(const __bf16* row, int kk, int hf) {
  union { v16bf v; v8bf h[2]; } u;
  u.h[0] = *(const v8bf*)(row + kk + hf * 8);
  u.h[1] = *(const v8bf*)(row + kk + 16 + hf * 8);
  return u.v;
}

static __device__ inline v16bf load_frag_f32_scaled(const float* row, int kk, int hf, float s) {
  union { v16bf v; __bf16 e[16]; } u;
  const float4 a0 = *(const float4*)(row + kk + hf * 8);
  const float4 a1 = *(const float4*)(row + kk + hf * 8 + 4);
  const float4 a2 = *(const float4*)(row + kk + 16 + hf * 8);
  const float4 a3 = *(const float4*)(row + kk + 16 + hf * 8 + 4);
  u.e[0] = (__bf16)(a0.x * s); u.e[1] = (__bf16)(a0.y * s);
  u.e[2] = (__bf16)(a0.z * s); u.e[3] = (__bf16)(a0.w * s);
  u.e[4] = (__bf16)(a1.x * s); u.e[5] = (__bf16)(a1.y * s);
  u.e[6] = (__bf16)(a1.z * s); u.e[7] = (__bf16)(a1.w * s);
  u.e[8] = (__bf16)(a2.x * s); u.e[9] = (__bf16)(a2.y * s);
  u.e[10] = (__bf16)(a2.z * s); u.e[11] = (__bf16)(a2.w * s);
  u.e[12] = (__bf16)(a3.x * s); u.e[13] = (__bf16)(a3.y * s);
  u.e[14] = (__bf16)(a3.z * s); u.e[15] = (__bf16)(a3.w * s);
  return u.v;
}

static __device__ inline unsigned long long umin64(unsigned long long a, unsigned long long b) {
  return a < b ? a : b;
}
static __device__ inline unsigned long long shfl_xor_u64(unsigned long long v, int m) {
  unsigned lo = (unsigned)(v & 0xFFFFFFFFull);
  unsigned hi = (unsigned)(v >> 32);
  lo = (unsigned)__shfl_xor((int)lo, m, 32);
  hi = (unsigned)__shfl_xor((int)hi, m, 32);
  return (((unsigned long long)hi) << 32) | lo;
}

static __device__ inline unsigned long long dist_key(float dist, int col) {
  unsigned u = __float_as_uint(dist);
  unsigned key32 = (u & 0x80000000u) ? ~u : (u | 0x80000000u);  // order-preserving map
  return (((unsigned long long)key32) << 32) | (unsigned)col;
}

// ---------- preprocessing ----------
__global__ void prep_weights(const float* Ws, const float* Wn, __bf16* WsT, __bf16* WnT) {
  long long i = (long long)blockIdx.x * blockDim.x + threadIdx.x;
  const long long total = (long long)KK * DD * DD;
  if (i >= total) return;
  int k = (int)(i / (DD * DD));
  int rem = (int)(i % (DD * DD));
  int r = rem / DD, c = rem % DD;
  size_t src = (size_t)k * DD * DD + (size_t)r * DD + c;
  size_t dstp = (size_t)k * DD * DD + (size_t)c * DD + r;
  WsT[dstp] = (__bf16)Ws[src];
  WnT[dstp] = (__bf16)Wn[src];
}

__global__ void prep_codebook(const float* cb, __bf16* cb_bf, float* cbn) {
  int row = blockIdx.x;  // 0 .. K*M-1
  int lane = threadIdx.x;
  const float* srow = cb + (size_t)row * DD;
  __bf16* drow = cb_bf + (size_t)row * DD;
  float s = 0.f;
  for (int d = lane; d < DD; d += 32) {
    float v = srow[d];
    drow[d] = (__bf16)v;
    s += v * v;
  }
#pragma unroll
  for (int m = 16; m >= 1; m >>= 1) s += __shfl_xor(s, m, 32);
  if (lane == 0) cbn[row] = s;
}

__global__ void prep_x(const float* X, __bf16* eprev, float* deg) {
  size_t stride = (size_t)gridDim.x * blockDim.x;
  size_t i0 = (size_t)blockIdx.x * blockDim.x + threadIdx.x;
  for (size_t i = i0; i < (size_t)NN * DD; i += stride) eprev[i] = (__bf16)X[i];
  for (size_t i = i0; i < (size_t)NN; i += stride) deg[i] = 0.f;
}

__global__ void edge_degree(const int* dst, float* deg) {
  int e = blockIdx.x * blockDim.x + threadIdx.x;
  if (e < EE) atomicAdd(&deg[dst[e]], 1.0f);
}

// ---------- per-hop ----------
__global__ void zero_hop(float* agg, unsigned long long* packed) {
  size_t stride = (size_t)gridDim.x * blockDim.x;
  size_t i0 = (size_t)blockIdx.x * blockDim.x + threadIdx.x;
  for (size_t i = i0; i < (size_t)NN * DD; i += stride) agg[i] = 0.f;
  for (size_t i = i0; i < (size_t)NN; i += stride) packed[i] = ~0ull;
}

// One thread per (edge, 8-float chunk): vector bf16 read, f32 atomic adds (L2-resident agg).
__global__ void scatter_add(const int* src, const int* dst, const __bf16* eprev, float* agg) {
  long long gid = (long long)blockIdx.x * blockDim.x + threadIdx.x;
  if (gid >= (long long)EE * (DD / 8)) return;
  int e = (int)(gid >> 6);
  int c = ((int)gid & 63) * 8;
  int s = src[e], d = dst[e];
  v8bf v = *(const v8bf*)(eprev + (size_t)s * DD + c);
  float* q = agg + (size_t)d * DD + c;
#pragma unroll
  for (int i = 0; i < 8; ++i) atomicAdd(q + i, (float)v[i]);
}

// h = relu(e_prev@Ws + (agg/deg)@Wn + b), output bf16. Wave computes 16x64.
// grid (N/16, 2), block 128 (4 waves -> 256 cols / WG).
__global__ void gemm_layer(const __bf16* Aprev, const float* agg, const float* deg,
                           const __bf16* WsT, const __bf16* WnT, const float* bias,
                           __bf16* H) {
  int wave = threadIdx.x >> 5;
  int lane = threadIdx.x & 31;
  int hf = lane >> 4, lr = lane & 15;
  int row0 = blockIdx.x * 16;
  int col0 = (blockIdx.y * 4 + wave) * 64;
  int r = row0 + lr;
  float rdeg = 1.0f / fmaxf(deg[r], 1.0f);
  const __bf16* arow = Aprev + (size_t)r * DD;
  const float* grow = agg + (size_t)r * DD;

  v8f acc[4] = {v8f{}, v8f{}, v8f{}, v8f{}};
  for (int kk = 0; kk < DD; kk += 32) {
    v16bf a_s = load_frag_row(arow, kk, hf);
    v16bf a_n = load_frag_f32_scaled(grow, kk, hf, rdeg);
#pragma unroll
    for (int t = 0; t < 4; ++t) {
      int bc = col0 + t * 16 + lr;
      v16bf bs = load_frag_row(WsT + (size_t)bc * DD, kk, hf);
      v16bf bn = load_frag_row(WnT + (size_t)bc * DD, kk, hf);
      acc[t] = __builtin_amdgcn_wmma_f32_16x16x32_bf16(false, a_s, false, bs,
                                                       (short)0, acc[t], false, false);
      acc[t] = __builtin_amdgcn_wmma_f32_16x16x32_bf16(false, a_n, false, bn,
                                                       (short)0, acc[t], false, false);
    }
  }
#pragma unroll
  for (int t = 0; t < 4; ++t) {
    int col = col0 + t * 16 + lr;
    float bc = bias[col];
#pragma unroll
    for (int i = 0; i < 8; ++i) {
      int rr = row0 + hf * 8 + i;
      float v = fmaxf(acc[t][i] + bc, 0.f);
      H[(size_t)rr * DD + col] = (__bf16)v;
    }
  }
}

// scores S = h@cb^T via WMMA; dist = ||cb||^2 - 2S; fused argmin via packed u64 atomicMin.
// Block = 256 threads (8 waves) covers 32 rows x 512 cols; the 32x512 bf16 A-panel is
// staged once in LDS (async-to-LDS path when available) and shared by all 8 waves.
// grid (N/32, 8).
__global__ void __launch_bounds__(256) gemm_assign(const __bf16* H, const __bf16* cb_bf,
                                                   const float* cbn,
                                                   unsigned long long* packed) {
  __shared__ __align__(16) __bf16 As[32 * DD];  // 32 KB
  int tid = threadIdx.x;
  int wave = tid >> 5;
  int lane = tid & 31;
  int hf = lane >> 4, lr = lane & 15;
  int row0 = blockIdx.x * 32;
  int colBase = (blockIdx.y * 8 + wave) * 64;

  // ---- stage A panel: 32 rows x 512 bf16 = 2048 x 16B chunks / 256 threads ----
  {
    const __bf16* gsrc = H + (size_t)row0 * DD;
#if HAVE_ASYNC_LDS
    for (int i = tid; i < 32 * 64; i += 256) {
      int rr = i >> 6;
      int cc = (i & 63) << 3;
      __builtin_amdgcn_global_load_async_to_lds_b128(
          (v4i*)(gsrc + (size_t)rr * DD + cc),
          (v4i*)&As[rr * DD + cc], 0, 0);
    }
    wait_asynccnt0();
#else
    for (int i = tid; i < 32 * 64; i += 256) {
      int rr = i >> 6;
      int cc = (i & 63) << 3;
      *(v8bf*)&As[rr * DD + cc] = *(const v8bf*)(gsrc + (size_t)rr * DD + cc);
    }
#endif
  }
  __syncthreads();

  v8f acc0[4] = {v8f{}, v8f{}, v8f{}, v8f{}};   // rows 0..15
  v8f acc1[4] = {v8f{}, v8f{}, v8f{}, v8f{}};   // rows 16..31
  for (int kk = 0; kk < DD; kk += 32) {
    v16bf a0 = load_frag_row(As + (size_t)lr * DD, kk, hf);          // rows 0..15
    v16bf a1 = load_frag_row(As + (size_t)(16 + lr) * DD, kk, hf);   // rows 16..31
#pragma unroll
    for (int t = 0; t < 4; ++t) {
      const __bf16* brow = cb_bf + (size_t)(colBase + t * 16 + lr) * DD;
      if (kk + 64 < DD) __builtin_prefetch(brow + kk + 64, 0, 1);
      v16bf b = load_frag_row(brow, kk, hf);
      acc0[t] = __builtin_amdgcn_wmma_f32_16x16x32_bf16(false, a0, false, b,
                                                        (short)0, acc0[t], false, false);
      acc1[t] = __builtin_amdgcn_wmma_f32_16x16x32_bf16(false, a1, false, b,
                                                        (short)0, acc1[t], false, false);
    }
  }

  unsigned long long best0[8], best1[8];
#pragma unroll
  for (int i = 0; i < 8; ++i) { best0[i] = ~0ull; best1[i] = ~0ull; }
#pragma unroll
  for (int t = 0; t < 4; ++t) {
    int col = colBase + t * 16 + lr;
    float cn = cbn[col];
#pragma unroll
    for (int i = 0; i < 8; ++i) {
      best0[i] = umin64(best0[i], dist_key(cn - 2.0f * acc0[t][i], col));
      best1[i] = umin64(best1[i], dist_key(cn - 2.0f * acc1[t][i], col));
    }
  }
#pragma unroll
  for (int m = 1; m < 16; m <<= 1) {
#pragma unroll
    for (int i = 0; i < 8; ++i) {
      best0[i] = umin64(best0[i], shfl_xor_u64(best0[i], m));
      best1[i] = umin64(best1[i], shfl_xor_u64(best1[i], m));
    }
  }
  if (lr == 0) {
#pragma unroll
    for (int i = 0; i < 8; ++i) {
      atomicMin(&packed[row0 + hf * 8 + i], best0[i]);
      atomicMin(&packed[row0 + 16 + hf * 8 + i], best1[i]);
    }
  }
}

// Extract winner idx, write token, gather codebook row -> next e_prev (and final f32 out).
__global__ void finalize_hop(const unsigned long long* packed, const float* cb_f32,
                             __bf16* eprev, float* out_e, float* out_z, int isLast) {
  int n = blockIdx.x;
  unsigned idx = (unsigned)(packed[n] & 0xFFFFFFFFull);
  if (threadIdx.x == 0) out_z[n] = (float)idx;
  const float* srow = cb_f32 + (size_t)idx * DD;
  for (int d = threadIdx.x; d < DD; d += blockDim.x) {
    float v = srow[d];
    eprev[(size_t)n * DD + d] = (__bf16)v;
    if (isLast) out_e[(size_t)n * DD + d] = v;
  }
}

extern "C" void kernel_launch(void* const* d_in, const int* in_sizes, int n_in,
                              void* d_out, int out_size, void* d_ws, size_t ws_size,
                              hipStream_t stream) {
  (void)in_sizes; (void)n_in; (void)out_size; (void)ws_size;
  const float* X  = (const float*)d_in[0];
  const int*   ei = (const int*)d_in[1];
  const float* CB = (const float*)d_in[2];
  const float* Ws = (const float*)d_in[3];
  const float* Wn = (const float*)d_in[4];
  const float* Bb = (const float*)d_in[5];
  const int* src = ei;
  const int* dst = ei + EE;

  char* p = (char*)d_ws;
  auto take = [&](size_t bytes) { char* r = p; p += (bytes + 255) & ~(size_t)255; return r; };
  __bf16* WsT   = (__bf16*)take((size_t)KK * DD * DD * 2);
  __bf16* WnT   = (__bf16*)take((size_t)KK * DD * DD * 2);
  __bf16* CBbf  = (__bf16*)take((size_t)KK * MM * DD * 2);
  float*  cbn   = (float*)take((size_t)KK * MM * 4);
  float*  deg   = (float*)take((size_t)NN * 4);
  float*  agg   = (float*)take((size_t)NN * DD * 4);
  __bf16* eprev = (__bf16*)take((size_t)NN * DD * 2);
  __bf16* Hbuf  = (__bf16*)take((size_t)NN * DD * 2);
  unsigned long long* packed = (unsigned long long*)take((size_t)NN * 8);

  float* out_e = (float*)d_out;
  float* out_z_base = (float*)d_out + (size_t)NN * DD;

  {
    long long total = (long long)KK * DD * DD;
    prep_weights<<<(unsigned)((total + 255) / 256), 256, 0, stream>>>(Ws, Wn, WsT, WnT);
    prep_codebook<<<KK * MM, 32, 0, stream>>>(CB, CBbf, cbn);
    prep_x<<<4096, 256, 0, stream>>>(X, eprev, deg);
    edge_degree<<<(EE + 255) / 256, 256, 0, stream>>>(dst, deg);
  }

  for (int k = 0; k < KK; ++k) {
    zero_hop<<<4096, 256, 0, stream>>>(agg, packed);
    long long sthreads = (long long)EE * (DD / 8);
    scatter_add<<<(unsigned)((sthreads + 255) / 256), 256, 0, stream>>>(src, dst, eprev, agg);
    gemm_layer<<<dim3(NN / 16, 2), 128, 0, stream>>>(
        eprev, agg, deg,
        WsT + (size_t)k * DD * DD, WnT + (size_t)k * DD * DD, Bb + (size_t)k * DD, Hbuf);
    gemm_assign<<<dim3(NN / 32, 8), 256, 0, stream>>>(
        Hbuf, CBbf + (size_t)k * MM * DD, cbn + (size_t)k * MM, packed);
    finalize_hop<<<NN, 128, 0, stream>>>(
        packed, CB + (size_t)k * MM * DD, eprev,
        out_e, out_z_base + (size_t)k * NN, (k == KK - 1) ? 1 : 0);
  }
}
